// QBertLMPredictionHead_42838003810885
// MI455X (gfx1250) — matrix operand order repaired and checked by reading, not compiled, not run on hardware
//
#include <hip/hip_runtime.h>
#include <hip/hip_bf16.h>
#include <math.h>

// ---------- types ----------
typedef __attribute__((ext_vector_type(16))) _Float16 v16h;
typedef __attribute__((ext_vector_type(8)))  _Float16 half8;
typedef __attribute__((ext_vector_type(4)))  _Float16 half4;
typedef __attribute__((ext_vector_type(8)))  float    v8f;
typedef __attribute__((ext_vector_type(4)))  float    f4;
typedef __attribute__((ext_vector_type(4)))  int      v4i;

#define HIDDEN   768
#define DIM      1024
#define NQ       10
#define NL       2
#define NANG     20
#define ROWS     4096   // 8*512
#define VOCAB    30522

// Block tile: 128(M) x 128(N), K-step 32, 256 threads (8 waves).
// Each wave computes a 32x64 sub-tile = 2(M) x 4(N) WMMA fragments.
#define BM 128
#define BN 128
#define BK 32
#define LDS_STRIDE 40   // 40 halves = 80 bytes (multiple of 16B -> aligned ds_load_b128)

#if __has_builtin(__builtin_amdgcn_global_load_async_to_lds_b128) && \
    __has_builtin(__builtin_amdgcn_s_wait_asynccnt)
#define USE_ASYNC_LDS 1
#else
#define USE_ASYNC_LDS 0
#endif

#if USE_ASYNC_LDS
typedef __attribute__((address_space(3))) v4i* lds_v4i_t;
#endif

__device__ __forceinline__ v16h load_frag_a(const _Float16* rowbase, int lhalf) {
    // A 16x32 f16 fragment, lane = 16*lhalf + m:
    //   elements 0..7  = K (lhalf*8)..(lhalf*8+7)
    //   elements 8..15 = K (16+lhalf*8)..(16+lhalf*8+7)
    const int khalf = lhalf * 8;
    half8 lo = *(const half8*)(rowbase + khalf);
    half8 hi = *(const half8*)(rowbase + khalf + 16);
    v16h r;
#pragma unroll
    for (int e = 0; e < 8; ++e) { r[e] = lo[e]; r[e + 8] = hi[e]; }
    return r;
}

__device__ __forceinline__ v16h load_frag_b(const _Float16* colbase, int lhalf) {
    // B 32x16 f16 fragment from transposed LDS tile Bls[n][k]:
    //   lane = 16*lhalf + n: elements 0..7 = K lhalf*8+e, elements 8..15 = K 16+lhalf*8+e
    const int kbase = lhalf * 8;
    half8 lo = *(const half8*)(colbase + kbase);
    half8 hi = *(const half8*)(colbase + kbase + 16);
    v16h r;
#pragma unroll
    for (int e = 0; e < 8; ++e) { r[e] = lo[e]; r[e + 8] = hi[e]; }
    return r;
}

__device__ __forceinline__ float gelu_f(float x) {
    return 0.5f * x * (1.0f + erff(x * 0.70710678118654752f));
}

// f16 x f16 -> f32 GEMM:  C = epi(A(MxK) @ BT(NxK)^T + bias)
// Staging is a pure 16B copy: async global->LDS (ASYNCcnt) when the toolchain
// exposes the gfx1250 builtins, else register-staged b128 copies.
// LDS double-buffered: one barrier per k-step. Output uses non-temporal
// stores so the L2-resident weight panel survives the 500MB output stream.
__global__ __launch_bounds__(256)
void gemm_wmma_f16_kernel(const _Float16* __restrict__ A,   // [M][K] f16
                          const _Float16* __restrict__ BT,  // [N][K] f16
                          const float* __restrict__ bias,
                          float* __restrict__ C,
                          int M, int N, int K, int do_gelu) {
    __shared__ _Float16 Als[2][BM][LDS_STRIDE];   // [buf][m][k]
    __shared__ _Float16 Bls[2][BN][LDS_STRIDE];   // [buf][n][k]

    const int tid  = threadIdx.x;
    const int m0   = blockIdx.y * BM;
    const int n0   = blockIdx.x * BN;

    const int w     = tid >> 5;       // wave 0..7
    const int lane  = tid & 31;
    const int wm    = w >> 1;         // 0..3 : 32-row band
    const int wn    = w & 1;          // 0..1 : 64-col band
    const int lrow  = lane & 15;      // fragment row/col within 16
    const int lhalf = lane >> 4;      // K-half select

#if USE_ASYNC_LDS
    // Async DMA-style staging: 512 half8-slots per tile, 4 per thread.
    auto async_load_tiles = [&](int k0, int buf) {
#pragma unroll
        for (int i = 0; i < 2; ++i) {
            int s8 = tid + i * 256;
            int m  = s8 >> 2;
            int k8 = (s8 & 3) << 3;
            __builtin_amdgcn_global_load_async_to_lds_b128(
                (v4i*)(A + (size_t)(m0 + m) * K + (k0 + k8)),
                (lds_v4i_t)&Als[buf][m][k8], 0, 0);
        }
#pragma unroll
        for (int i = 0; i < 2; ++i) {
            int s8 = tid + i * 256;
            int n  = s8 >> 2;
            int k8 = (s8 & 3) << 3;
            int gn = n0 + n; if (gn >= N) gn = N - 1;   // clamp; cols >= N discarded
            __builtin_amdgcn_global_load_async_to_lds_b128(
                (v4i*)(BT + (size_t)gn * K + (k0 + k8)),
                (lds_v4i_t)&Bls[buf][n][k8], 0, 0);
        }
    };
#else
    half8 aR[2], bR[2];
    auto load_tiles = [&](int k0) {
#pragma unroll
        for (int i = 0; i < 2; ++i) {
            int s8 = tid + i * 256;
            int m  = s8 >> 2;
            int k8 = (s8 & 3) << 3;
            aR[i] = *(const half8*)(A + (size_t)(m0 + m) * K + (k0 + k8));
        }
#pragma unroll
        for (int i = 0; i < 2; ++i) {
            int s8 = tid + i * 256;
            int n  = s8 >> 2;
            int k8 = (s8 & 3) << 3;
            int gn = n0 + n;
            bR[i] = (gn < N) ? *(const half8*)(BT + (size_t)gn * K + (k0 + k8))
                             : (half8)(_Float16)0.0f;
        }
    };
    auto store_tiles = [&](int buf) {
#pragma unroll
        for (int i = 0; i < 2; ++i) {
            int s8 = tid + i * 256;
            int m  = s8 >> 2;
            int k8 = (s8 & 3) << 3;
            *(half8*)&Als[buf][m][k8] = aR[i];               // ds_store_b128
        }
#pragma unroll
        for (int i = 0; i < 2; ++i) {
            int s8 = tid + i * 256;
            int n  = s8 >> 2;
            int k8 = (s8 & 3) << 3;
            *(half8*)&Bls[buf][n][k8] = bR[i];               // ds_store_b128
        }
    };
#endif

    v8f acc[2][4];
#pragma unroll
    for (int i = 0; i < 2; ++i)
#pragma unroll
        for (int j = 0; j < 4; ++j) acc[i][j] = (v8f)(0.0f);

    // Prologue: fill buffer 0.
#if USE_ASYNC_LDS
    async_load_tiles(0, 0);
    __builtin_amdgcn_s_wait_asynccnt(0);
#else
    load_tiles(0);
    store_tiles(0);
#endif
    __syncthreads();

    const int nsteps = K / BK;
    for (int s = 0; s < nsteps; ++s) {
        const int buf  = s & 1;
        const bool more = (s + 1) < nsteps;

#if USE_ASYNC_LDS
        if (more) async_load_tiles((s + 1) * BK, buf ^ 1);  // DMA behind the WMMAs
#else
        if (more) load_tiles((s + 1) * BK);   // global prefetch in flight over WMMAs
#endif

        v16h av[2], bv[4];
#pragma unroll
        for (int i = 0; i < 2; ++i)
            av[i] = load_frag_a(&Als[buf][wm * 32 + i * 16 + lrow][0], lhalf);
#pragma unroll
        for (int j = 0; j < 4; ++j)
            bv[j] = load_frag_b(&Bls[buf][wn * 64 + j * 16 + lrow][0], lhalf);

#pragma unroll
        for (int i = 0; i < 2; ++i)
#pragma unroll
            for (int j = 0; j < 4; ++j)
                acc[i][j] = __builtin_amdgcn_wmma_f32_16x16x32_f16(
                    false, av[i], false, bv[j], (short)0, acc[i][j], false, false);

#if USE_ASYNC_LDS
        if (more) __builtin_amdgcn_s_wait_asynccnt(0);
#else
        if (more) store_tiles(buf ^ 1);       // fill other buffer behind the WMMAs
#endif
        __syncthreads();                      // single barrier per k-step
    }

    // ---- epilogue: bias (+ GELU), guarded non-temporal store ----
#pragma unroll
    for (int i = 0; i < 2; ++i) {
#pragma unroll
        for (int j = 0; j < 4; ++j) {
            int col = n0 + wn * 64 + j * 16 + lrow;
            if (col < N) {
                float b = bias[col];
#pragma unroll
                for (int v = 0; v < 8; ++v) {
                    int row = m0 + wm * 32 + i * 16 + v + 8 * lhalf;
                    float x = acc[i][j][v] + b;
                    if (do_gelu) x = gelu_f(x);
                    __builtin_nontemporal_store(x, &C[(size_t)row * N + col]);
                }
            }
        }
    }
}

// Tiled transpose + f32->f16 convert:  in[K][N] f32  ->  outT[N][K] f16
__global__ __launch_bounds__(256)
void transpose_cvt_kernel(const float* __restrict__ in, _Float16* __restrict__ outT,
                          int K, int N) {
    __shared__ float t[32][33];
    const int k0 = blockIdx.y * 32;
    const int n0 = blockIdx.x * 32;
    const int tx = threadIdx.x & 31;
    const int ty = threadIdx.x >> 5;   // 0..7
#pragma unroll
    for (int r = 0; r < 32; r += 8) {
        int k = k0 + ty + r, n = n0 + tx;
        t[ty + r][tx] = (k < K && n < N) ? in[(size_t)k * N + n] : 0.0f;
    }
    __syncthreads();
#pragma unroll
    for (int r = 0; r < 32; r += 8) {
        int n = n0 + ty + r, k = k0 + tx;
        if (n < N && k < K) outT[(size_t)n * K + k] = (_Float16)t[tx][ty + r];
    }
}

// Elementwise f32 -> f16 (n multiple of 4)
__global__ __launch_bounds__(256)
void cvt_f16_kernel(const float* __restrict__ in, _Float16* __restrict__ out, int n4) {
    int i = (blockIdx.x * 256 + threadIdx.x);
    if (i < n4) {
        f4 v = *(const f4*)(in + (size_t)i * 4);
        half4 h;
#pragma unroll
        for (int e = 0; e < 4; ++e) h[e] = (_Float16)v[e];
        *(half4*)(out + (size_t)i * 4) = h;
    }
}

// LayerNorm(row of 768) -> 20 dots with ent_w -> sigmoid*2pi
__global__ __launch_bounds__(256)
void ln_ent_kernel(const float* __restrict__ h, const float* __restrict__ ln_w,
                   const float* __restrict__ ln_b, const float* __restrict__ ent_w,
                   const float* __restrict__ ent_b, float* __restrict__ angles) {
    __shared__ float red[256];
    __shared__ float hn[HIDDEN];
    const int row = blockIdx.x;
    const int tid = threadIdx.x;
    const float* hr = h + (size_t)row * HIDDEN;

    float x0 = hr[tid], x1 = hr[tid + 256], x2 = hr[tid + 512];

    red[tid] = x0 + x1 + x2;
    __syncthreads();
    for (int off = 128; off > 0; off >>= 1) {
        if (tid < off) red[tid] += red[tid + off];
        __syncthreads();
    }
    float mean = red[0] * (1.0f / HIDDEN);
    __syncthreads();

    float d0 = x0 - mean, d1 = x1 - mean, d2 = x2 - mean;
    red[tid] = d0 * d0 + d1 * d1 + d2 * d2;
    __syncthreads();
    for (int off = 128; off > 0; off >>= 1) {
        if (tid < off) red[tid] += red[tid + off];
        __syncthreads();
    }
    float inv = rsqrtf(red[0] * (1.0f / HIDDEN) + 1e-12f);

    hn[tid]       = ln_w[tid]       * (d0 * inv) + ln_b[tid];
    hn[tid + 256] = ln_w[tid + 256] * (d1 * inv) + ln_b[tid + 256];
    hn[tid + 512] = ln_w[tid + 512] * (d2 * inv) + ln_b[tid + 512];
    __syncthreads();

    const int w = tid >> 5, lane = tid & 31;
    for (int o = w; o < NANG; o += 8) {
        float acc = 0.0f;
        for (int i = lane; i < HIDDEN; i += 32) acc += hn[i] * ent_w[i * NANG + o];
#pragma unroll
        for (int off = 16; off > 0; off >>= 1) acc += __shfl_down(acc, off, 32);
        if (lane == 0) {
            float s = 1.0f / (1.0f + expf(-(acc + ent_b[o])));
            angles[(size_t)row * NANG + o] = 6.283185307179586f * s;
        }
    }
}

// 10-qubit state-vector simulation in LDS, one row per block (512 threads).
// Emits probabilities directly in f16 (values in [0,1]) for the vocab GEMM.
__global__ __launch_bounds__(512)
void quantum_kernel(const float* __restrict__ angles, _Float16* __restrict__ probs) {
    __shared__ float re[DIM];
    __shared__ float im[DIM];
    __shared__ float th[NANG];
    const int row = blockIdx.x;
    const int tid = threadIdx.x;

    if (tid < NANG) th[tid] = angles[(size_t)row * NANG + tid];
    re[tid] = 0.0f; re[tid + 512] = 0.0f;
    im[tid] = 0.0f; im[tid + 512] = 0.0f;
    __syncthreads();
    if (tid == 0) re[0] = 1.0f;
    __syncthreads();

    for (int l = 0; l < NL; ++l) {
        // RX gates
        for (int q = 0; q < NQ; ++q) {
            float half_t = th[l * NQ + q] * 0.5f;
            float c = cosf(half_t), s = sinf(half_t);
            int sh = (NQ - 1) - q;                        // pair stride log2
            int stride = 1 << sh;
            int i0 = ((tid >> sh) << (sh + 1)) | (tid & (stride - 1));
            int i1 = i0 + stride;
            float r0 = re[i0], m0 = im[i0], r1 = re[i1], m1 = im[i1];
            re[i0] = c * r0 + s * m1;   im[i0] = c * m0 - s * r1;  // b0 = c*a0 - i*s*a1
            re[i1] = c * r1 + s * m0;   im[i1] = c * m1 - s * r0;  // b1 = c*a1 - i*s*a0
            __syncthreads();
        }
        // CNOT ring: new[j] = old[j ^ (ctrl_bit(j) << (9 - tq))]
        for (int q = 0; q < NQ; ++q) {
            int tq = (q + 1) % NQ;
            int j0 = tid, j1 = tid + 512;
            int s0 = j0 ^ ((((unsigned)j0 >> (9 - q)) & 1u) << (9 - tq));
            int s1 = j1 ^ ((((unsigned)j1 >> (9 - q)) & 1u) << (9 - tq));
            float a = re[s0], b = im[s0], c = re[s1], d = im[s1];
            __syncthreads();
            re[j0] = a; im[j0] = b; re[j1] = c; im[j1] = d;
            __syncthreads();
        }
    }

    probs[(size_t)row * DIM + tid] =
        (_Float16)(re[tid] * re[tid] + im[tid] * im[tid]);
    probs[(size_t)row * DIM + tid + 512] =
        (_Float16)(re[tid + 512] * re[tid + 512] + im[tid + 512] * im[tid + 512]);
}

extern "C" void kernel_launch(void* const* d_in, const int* in_sizes, int n_in,
                              void* d_out, int out_size, void* d_ws, size_t ws_size,
                              hipStream_t stream) {
    const float* hidden  = (const float*)d_in[0];  // (4096, 768)
    const float* dense_w = (const float*)d_in[1];  // (768, 768)
    const float* dense_b = (const float*)d_in[2];  // (768)
    const float* ln_w    = (const float*)d_in[3];  // (768)
    const float* ln_b    = (const float*)d_in[4];  // (768)
    const float* ent_w   = (const float*)d_in[5];  // (768, 20)
    const float* ent_b   = (const float*)d_in[6];  // (20)
    const float* proj_w  = (const float*)d_in[7];  // (1024, 30522)
    const float* proj_b  = (const float*)d_in[8];  // (30522)
    float* out = (float*)d_out;                    // (4096, 30522)

    // ---- workspace layout ----
    char* ws = (char*)d_ws;
    size_t off = 0;
    _Float16* pwT_h   = (_Float16*)(ws + off); off += (size_t)VOCAB * DIM * 2;     // 62.5MB
    float*    h_ws    = (float*)   (ws + off); off += (size_t)ROWS * HIDDEN * 4;   // 12.6MB
    float*    ang_ws  = (float*)   (ws + off); off += (size_t)ROWS * NANG * 4;     // 0.33MB
    _Float16* probs_h = (_Float16*)(ws + off); off += (size_t)ROWS * DIM * 2;      // 8.4MB
    _Float16* hid_h   = (_Float16*)(ws + off); off += (size_t)ROWS * HIDDEN * 2;   // 6.3MB
    _Float16* dwT_h   = (_Float16*)(ws + off); off += (size_t)HIDDEN * HIDDEN * 2; // 1.2MB

    // Pre-pass: convert / transpose operands to f16 once.
    cvt_f16_kernel<<<(ROWS * HIDDEN / 4 + 255) / 256, 256, 0, stream>>>(
        hidden, hid_h, ROWS * HIDDEN / 4);
    transpose_cvt_kernel<<<dim3((HIDDEN + 31) / 32, (HIDDEN + 31) / 32), 256, 0, stream>>>(
        dense_w, dwT_h, HIDDEN, HIDDEN);
    transpose_cvt_kernel<<<dim3((VOCAB + 31) / 32, (DIM + 31) / 32), 256, 0, stream>>>(
        proj_w, pwT_h, DIM, VOCAB);

    // Stage 1: h = gelu(hidden @ dense_w + dense_b)
    {
        dim3 grid((HIDDEN + BN - 1) / BN, ROWS / BM);   // (6, 32)
        gemm_wmma_f16_kernel<<<grid, 256, 0, stream>>>(hid_h, dwT_h, dense_b, h_ws,
                                                       ROWS, HIDDEN, HIDDEN, /*gelu=*/1);
    }
    // Stage 2: layernorm + entangler head -> angles
    ln_ent_kernel<<<ROWS, 256, 0, stream>>>(h_ws, ln_w, ln_b, ent_w, ent_b, ang_ws);

    // Stage 3: quantum circuit simulation -> probs (f16)
    quantum_kernel<<<ROWS, 512, 0, stream>>>(ang_ws, probs_h);

    // Stage 4: out = probs @ proj_w + proj_b
    {
        dim3 grid((VOCAB + BN - 1) / BN, ROWS / BM);    // (239, 32)
        gemm_wmma_f16_kernel<<<grid, 256, 0, stream>>>(probs_h, pwT_h, proj_b, out,
                                                       ROWS, VOCAB, DIM, /*gelu=*/0);
    }
}
